// CostVolumeAttention11_66546223284241
// MI455X (gfx1250) — matrix operand order repaired and checked by reading, not compiled, not run on hardware
//
#include <hip/hip_runtime.h>
#include <hip/hip_bf16.h>

typedef __bf16 v16bf __attribute__((ext_vector_type(16)));
typedef float  v8f   __attribute__((ext_vector_type(8)));

#define NH 8
#define HD 32
#define TT 8
#define HH 32
#define WW 32
#define PP 1024   // H*W
#define CC 256
#define NN 8192
#define BB 8

__device__ __forceinline__ v16bf bf_zero16() {
  v16bf z;
#pragma unroll
  for (int i = 0; i < 16; ++i) z[i] = (__bf16)0.0f;
  return z;
}

// A-fragment (16x32, bf16): per ISA, lane-half hf holds K in {hf*8..hf*8+7, 16+hf*8..16+hf*8+7}
__device__ __forceinline__ v16bf load_a_f32(const float* p, int hf) {
  v16bf a;
  const float* p0 = p + hf * 8;
  const float* p1 = p + 16 + hf * 8;
#pragma unroll
  for (int i = 0; i < 8; ++i) { a[i] = (__bf16)p0[i]; a[i + 8] = (__bf16)p1[i]; }
  return a;
}

__device__ __forceinline__ v16bf load_a_bf16(const __bf16* p, int hf) {
  v16bf a;
  const __bf16* p0 = p + hf * 8;
  const __bf16* p1 = p + 16 + hf * 8;
#pragma unroll
  for (int i = 0; i < 8; ++i) { a[i] = p0[i]; a[i + 8] = p1[i]; }
  return a;
}

// B-fragment (32x16, bf16): lane-half hf holds K in {hf*8..hf*8+7, 16+hf*8..16+hf*8+7} of one column
__device__ __forceinline__ v16bf load_b_bf16(const __bf16* p, int hf) {
  v16bf b;
  const __bf16* p0 = p + hf * 8;
  const __bf16* p1 = p + 16 + hf * 8;
#pragma unroll
  for (int i = 0; i < 8; ++i) { b[i] = p0[i]; b[i + 8] = p1[i]; }
  return b;
}

__device__ __forceinline__ v8f wmma_bf16(v16bf a, v16bf b, v8f c) {
  return __builtin_amdgcn_wmma_f32_16x16x32_bf16(false, a, false, b, (short)0, c,
                                                 false, false);
}

// ---------------------------------------------------------------------------
// Kernel 0: convert weights to bf16; w_corr transposed + padded to [32][64]
// ---------------------------------------------------------------------------
__global__ void cva_cvt_weights(const float* __restrict__ wqk,
                                const float* __restrict__ wproj,
                                const float* __restrict__ wcorr,
                                __bf16* __restrict__ wqk_bf,
                                __bf16* __restrict__ wproj_bf,
                                __bf16* __restrict__ wcorrT) {
  int i = blockIdx.x * 256 + threadIdx.x;
  if (i < 512 * 256) wqk_bf[i] = (__bf16)wqk[i];
  if (i < 256 * 256) wproj_bf[i] = (__bf16)wproj[i];
  if (i < 32 * 64) {
    int d = i >> 6, k = i & 63;
    float v = (k < 49) ? wcorr[d * 49 + k] : 0.0f;
    wcorrT[i] = (__bf16)v;
  }
}

// ---------------------------------------------------------------------------
// Kernel 1: qk = x @ w_qk^T  (bf16 WMMA, f32 accum), fused L2-normalize over
// hd=32 chunks, scatter to qn / time-shifted kn in bf16.
// Wave tile: 16 rows x 64 cols (2 heads). Grid: 4096 row-tiles x 8 col-blocks.
// ---------------------------------------------------------------------------
__global__ void __launch_bounds__(128) cva_qk_norm(const float* __restrict__ x,
                                                   const __bf16* __restrict__ wqk_bf,
                                                   __bf16* __restrict__ qn,
                                                   __bf16* __restrict__ kn) {
  __shared__ float tile[4][16][65];
  const int wv = threadIdx.x >> 5;
  const int lane = threadIdx.x & 31;
  const int r = lane & 15, hf = lane >> 4;
  const unsigned w = blockIdx.x * 4 + wv;
  const int cb64 = w & 7;
  const int rt = w >> 3;
  const int row0 = rt * 16;
  const int d0 = cb64 * 64;

  v8f z8 = {0.f, 0.f, 0.f, 0.f, 0.f, 0.f, 0.f, 0.f};
  v8f c0 = z8, c1 = z8, c2 = z8, c3 = z8;

  const float* arow = x + (size_t)(row0 + r) * CC;
  for (int kk = 0; kk < CC; kk += 32) {
    __builtin_prefetch(arow + ((kk + 32) & (CC - 1)), 0, 0);
    v16bf a = load_a_f32(arow + kk, hf);
    v16bf b0 = load_b_bf16(wqk_bf + (size_t)(d0 + 0 + r) * CC + kk, hf);
    v16bf b1 = load_b_bf16(wqk_bf + (size_t)(d0 + 16 + r) * CC + kk, hf);
    v16bf b2 = load_b_bf16(wqk_bf + (size_t)(d0 + 32 + r) * CC + kk, hf);
    v16bf b3 = load_b_bf16(wqk_bf + (size_t)(d0 + 48 + r) * CC + kk, hf);
    c0 = wmma_bf16(a, b0, c0);
    c1 = wmma_bf16(a, b1, c1);
    c2 = wmma_bf16(a, b2, c2);
    c3 = wmma_bf16(a, b3, c3);
  }
#pragma unroll
  for (int v = 0; v < 8; ++v) {
    int M = v + 8 * hf;
    tile[wv][M][0 + r] = c0[v];
    tile[wv][M][16 + r] = c1[v];
    tile[wv][M][32 + r] = c2[v];
    tile[wv][M][48 + r] = c3[v];
  }
  __syncthreads();

  // lane -> (row rr, chunk ch): normalize 32 values, pack bf16 pairs
  const int rr = r, ch = hf;
  float ss = 0.f;
#pragma unroll
  for (int j = 0; j < 32; ++j) {
    float v = tile[wv][rr][ch * 32 + j];
    ss += v * v;
  }
  float inv = 1.0f / fmaxf(sqrtf(ss), 1e-12f);

  const int b = row0 / NN;
  const int ni = row0 % NN;
  const int tIdx = ni / PP;
  const int p = (ni % PP) + rr;
  const bool isK = (d0 >= CC);
  const int h = ((d0 % CC) / HD) + ch;
  const int bh = b * NH + h;

  unsigned pk[16];
#pragma unroll
  for (int j = 0; j < 16; ++j) {
    __bf16 lo = (__bf16)(tile[wv][rr][ch * 32 + 2 * j] * inv);
    __bf16 hi = (__bf16)(tile[wv][rr][ch * 32 + 2 * j + 1] * inv);
    unsigned short ulo = __builtin_bit_cast(unsigned short, lo);
    unsigned short uhi = __builtin_bit_cast(unsigned short, hi);
    pk[j] = (unsigned)ulo | ((unsigned)uhi << 16);
  }
  if (!isK) {
    unsigned* dst = (unsigned*)(qn + (((size_t)(bh * TT + tIdx)) * PP + p) * HD);
#pragma unroll
    for (int j = 0; j < 16; ++j) dst[j] = pk[j];
  } else {
    if (tIdx > 0) {
      unsigned* dst = (unsigned*)(kn + (((size_t)(bh * TT + (tIdx - 1))) * PP + p) * HD);
#pragma unroll
      for (int j = 0; j < 16; ++j) dst[j] = pk[j];
    }
    if (tIdx == TT - 1) {
      unsigned* dst = (unsigned*)(kn + (((size_t)(bh * TT + (TT - 1))) * PP + p) * HD);
#pragma unroll
      for (int j = 0; j < 16; ++j) dst[j] = pk[j];
    }
  }
}

// ---------------------------------------------------------------------------
// Kernel 2: 7x7 correlation as WMMA (per dy: 16x32 qn-tile x 32x22 kn-strip),
// diagonal-band extract via LDS, then 49->32 projection (2x K=32 WMMAs,
// seeded with b_corr). Output scattered to out_pre[b][n][c] in bf16.
// ---------------------------------------------------------------------------
__global__ void __launch_bounds__(128) cva_corr(const __bf16* __restrict__ qn,
                                                const __bf16* __restrict__ kn,
                                                const __bf16* __restrict__ wcorrT,
                                                const float* __restrict__ bcorr,
                                                __bf16* __restrict__ out_pre) {
  __shared__ float scratch[4][16][33];
  __shared__ float corrbuf[4][16][64];
  const int wv = threadIdx.x >> 5;
  const int lane = threadIdx.x & 31;
  const int r = lane & 15, hf = lane >> 4;
  const unsigned w = blockIdx.x * 4 + wv;
  const int xt = w & 1;
  const int y = (w >> 1) & 31;
  const int bt = w >> 6;
  const int x0 = xt * 16;

  for (int i = lane; i < 16 * 64; i += 32) corrbuf[wv][i >> 6][i & 63] = 0.f;

  const int pA = y * WW + x0 + r;
  v16bf a = load_a_bf16(qn + ((size_t)bt * PP + pA) * HD, hf);

  const v16bf zb = bf_zero16();
  const v8f z8 = {0.f, 0.f, 0.f, 0.f, 0.f, 0.f, 0.f, 0.f};

  for (int dy = 0; dy < 7; ++dy) {
    int yy = y + dy - 3;
    bool vrow = (yy >= 0) && (yy < HH);
    int xc0 = x0 - 3 + r;   // columns x0-3 .. x0+12
    int xc1 = x0 + 13 + r;  // columns x0+13 .. x0+28 (only first 6 used)
    v16bf b0 = zb, b1 = zb;
    if (vrow && xc0 >= 0 && xc0 < WW)
      b0 = load_b_bf16(kn + ((size_t)bt * PP + yy * WW + xc0) * HD, hf);
    if (vrow && xc1 < WW)
      b1 = load_b_bf16(kn + ((size_t)bt * PP + yy * WW + xc1) * HD, hf);
    v8f d0 = wmma_bf16(a, b0, z8);
    v8f d1 = wmma_bf16(a, b1, z8);
#pragma unroll
    for (int v = 0; v < 8; ++v) {
      int M = v + 8 * hf;
      scratch[wv][M][r] = d0[v];
      scratch[wv][M][16 + r] = d1[v];
    }
    __syncthreads();
    // corr[row][dy][dx] = D[row][row + dx + 3]
    for (int i = lane; i < 112; i += 32) {
      int rrow = i / 7, dxi = i % 7;
      corrbuf[wv][rrow][dy * 7 + dxi] = scratch[wv][rrow][rrow + dxi];
    }
    __syncthreads();
  }

  // v = corr(16x49, padded to 64) @ w_corr^T(49x32) + b_corr
  float bc0 = bcorr[r];
  float bc1 = bcorr[16 + r];
  v8f acc0 = {bc0, bc0, bc0, bc0, bc0, bc0, bc0, bc0};
  v8f acc1 = {bc1, bc1, bc1, bc1, bc1, bc1, bc1, bc1};
#pragma unroll
  for (int s = 0; s < 2; ++s) {
    v16bf a2;
    const float* cr = &corrbuf[wv][r][0];
#pragma unroll
    for (int i = 0; i < 8; ++i) {
      a2[i] = (__bf16)cr[s * 32 + hf * 8 + i];
      a2[i + 8] = (__bf16)cr[s * 32 + 16 + hf * 8 + i];
    }
    v16bf b20 = load_b_bf16(wcorrT + (size_t)(0 + r) * 64 + s * 32, hf);
    v16bf b21 = load_b_bf16(wcorrT + (size_t)(16 + r) * 64 + s * 32, hf);
    acc0 = wmma_bf16(a2, b20, acc0);
    acc1 = wmma_bf16(a2, b21, acc1);
  }

  const int b = bt >> 6;
  const int h = (bt >> 3) & 7;
  const int tIdx = bt & 7;
#pragma unroll
  for (int v = 0; v < 8; ++v) {
    int M = v + 8 * hf;
    int n = tIdx * PP + y * WW + x0 + M;
    size_t base = ((size_t)b * NN + n) * CC + h * HD;
    out_pre[base + r] = (__bf16)acc0[v];
    out_pre[base + 16 + r] = (__bf16)acc1[v];
  }
}

// ---------------------------------------------------------------------------
// Kernel 3: out = out_pre @ w_proj^T + b_proj  (bf16 WMMA, f32 out)
// ---------------------------------------------------------------------------
__global__ void __launch_bounds__(128) cva_proj(const __bf16* __restrict__ apre,
                                                const __bf16* __restrict__ wproj_bf,
                                                const float* __restrict__ bproj,
                                                float* __restrict__ out) {
  const int wv = threadIdx.x >> 5;
  const int lane = threadIdx.x & 31;
  const int r = lane & 15, hf = lane >> 4;
  const unsigned w = blockIdx.x * 4 + wv;
  const int cb64 = w & 3;
  const int rt = w >> 2;
  const int row0 = rt * 16;
  const int d0 = cb64 * 64;

  v8f z8 = {0.f, 0.f, 0.f, 0.f, 0.f, 0.f, 0.f, 0.f};
  v8f c0 = z8, c1 = z8, c2 = z8, c3 = z8;

  const __bf16* arow = apre + (size_t)(row0 + r) * CC;
  for (int kk = 0; kk < CC; kk += 32) {
    v16bf a = load_a_bf16(arow + kk, hf);
    v16bf b0 = load_b_bf16(wproj_bf + (size_t)(d0 + 0 + r) * CC + kk, hf);
    v16bf b1 = load_b_bf16(wproj_bf + (size_t)(d0 + 16 + r) * CC + kk, hf);
    v16bf b2 = load_b_bf16(wproj_bf + (size_t)(d0 + 32 + r) * CC + kk, hf);
    v16bf b3 = load_b_bf16(wproj_bf + (size_t)(d0 + 48 + r) * CC + kk, hf);
    c0 = wmma_bf16(a, b0, c0);
    c1 = wmma_bf16(a, b1, c1);
    c2 = wmma_bf16(a, b2, c2);
    c3 = wmma_bf16(a, b3, c3);
  }
  float bp0 = bproj[d0 + r];
  float bp1 = bproj[d0 + 16 + r];
  float bp2 = bproj[d0 + 32 + r];
  float bp3 = bproj[d0 + 48 + r];
#pragma unroll
  for (int v = 0; v < 8; ++v) {
    int M = v + 8 * hf;
    size_t base = (size_t)(row0 + M) * CC + d0;
    out[base + r] = c0[v] + bp0;
    out[base + 16 + r] = c1[v] + bp1;
    out[base + 32 + r] = c2[v] + bp2;
    out[base + 48 + r] = c3[v] + bp3;
  }
}

extern "C" void kernel_launch(void* const* d_in, const int* in_sizes, int n_in,
                              void* d_out, int out_size, void* d_ws, size_t ws_size,
                              hipStream_t stream) {
  const float* x = (const float*)d_in[0];
  const float* w_qk = (const float*)d_in[1];
  const float* w_corr = (const float*)d_in[2];
  const float* b_corr = (const float*)d_in[3];
  const float* w_proj = (const float*)d_in[4];
  const float* b_proj = (const float*)d_in[5];
  float* out = (float*)d_out;

  char* ws = (char*)d_ws;
  size_t off = 0;
  auto alloc = [&](size_t bytes) {
    char* p = ws + off;
    off = (off + bytes + 255) & ~(size_t)255;
    return p;
  };
  __bf16* wqk_bf = (__bf16*)alloc((size_t)512 * 256 * 2);
  __bf16* wproj_bf = (__bf16*)alloc((size_t)256 * 256 * 2);
  __bf16* wcorrT = (__bf16*)alloc((size_t)32 * 64 * 2);
  __bf16* qn = (__bf16*)alloc((size_t)512 * 1024 * 32 * 2);
  __bf16* kn = (__bf16*)alloc((size_t)512 * 1024 * 32 * 2);
  __bf16* out_pre = (__bf16*)alloc((size_t)8 * 8192 * 256 * 2);
  (void)in_sizes; (void)n_in; (void)out_size; (void)ws_size;

  cva_cvt_weights<<<512, 256, 0, stream>>>(w_qk, w_proj, w_corr, wqk_bf, wproj_bf, wcorrT);
  // 4096 row-tiles x 8 col-blocks = 32768 waves = 8192 blocks of 4 waves
  cva_qk_norm<<<8192, 128, 0, stream>>>(x, wqk_bf, qn, kn);
  // 512 bt x 32 rows x 2 x-tiles = 32768 waves
  cva_corr<<<8192, 128, 0, stream>>>(qn, kn, wcorrT, b_corr, out_pre);
  // 4096 row-tiles x 4 col-blocks = 16384 waves
  cva_proj<<<4096, 128, 0, stream>>>(out_pre, wproj_bf, b_proj, out);
}
